// VAEUnimodal_12232066859131
// MI455X (gfx1250) — compile-verified, matrix operand order checked by reading, not run on hardware
//
#include <hip/hip_runtime.h>

// ---------------------------------------------------------------------------
// VAE encoder+sampler for MI455X (gfx1250, wave32, WMMA).
// bf16 WMMA inputs + f32 accumulate; scalar/solve math in f32.
// ---------------------------------------------------------------------------

#define B_   64
#define S_   128
#define IN_  256
#define H_   256
#define NS_  4
#define DX_  48
#define DZ_  3
#define G_   16
#define ROWS_ (B_ * S_)          // 8192
#define EPSV  1e-5f

typedef __attribute__((ext_vector_type(16))) __bf16 v16bf;
typedef __attribute__((ext_vector_type(8)))  float  v8f;

#if defined(__HIP_DEVICE_COMPILE__) && \
    __has_builtin(__builtin_amdgcn_global_load_async_to_lds_b128) && \
    __has_builtin(__builtin_amdgcn_s_wait_asynccnt)
#define HAVE_ASYNC_LDS 1
// Builtin signature (from clang diagnostic): param0 is a 16B int-vector
// pointer in the global (AS1) address space; param1 is the LDS (AS3) side.
typedef int v4i_ __attribute__((vector_size(16)));
typedef __attribute__((address_space(1))) v4i_* as1_v4i_p;
typedef __attribute__((address_space(3))) v4i_* as3_v4i_p;
#endif

union ABFrag { v16bf v; unsigned u[8]; };

__device__ __forceinline__ unsigned short f2bf(float f) {
  unsigned x = __float_as_uint(f);
  unsigned r = x + 0x7FFFu + ((x >> 16) & 1u);   // round-to-nearest-even
  return (unsigned short)(r >> 16);
}

// Load a 16x32 bf16 fragment (A: rc=row base from [M][ld]; B: rc=col base from
// [N][ld], i.e. W row-major since every GEMM here is X*W^T).
// Per ISA 7.12.2: lane holds row/col (lane&15); VGPR v holds K =
// (v>=4?16:0) + (lane>>4)*8 + (v&3)*2 (two bf16 packed per dword).
// The two 16B-contiguous halves vectorize into b128 loads.
__device__ __forceinline__ void load_frag(ABFrag& f, const unsigned short* base,
                                          int rc, int ld, int k0, int lane) {
  const unsigned short* p = base + (size_t)(rc + (lane & 15)) * ld + k0 + (lane >> 4) * 8;
#pragma unroll
  for (int v = 0; v < 4; ++v) {
    f.u[v]     = *(const unsigned*)(p + 2 * v);
    f.u[v + 4] = *(const unsigned*)(p + 16 + 2 * v);
  }
}

__device__ __forceinline__ v8f wmma_bf16(const ABFrag& a, const ABFrag& b, v8f c) {
  return __builtin_amdgcn_wmma_f32_16x16x32_bf16(false, a.v, false, b.v,
                                                 (short)0, c, false, false);
}

// --------------------------- f32 -> bf16 convert ---------------------------
__global__ void k_cvt_bf16(const float* __restrict__ in,
                           unsigned short* __restrict__ out, int n) {
  int i = blockIdx.x * blockDim.x + threadIdx.x;
  if (i < n) out[i] = f2bf(in[i]);
}

// ---------------- input projection: pre = y*Wih^T + bih + bhh ---------------
// grid (512, 4 combos), block 256 (8 waves, each a 16x32 strip: 2 N-tiles
// sharing one A fragment -> half the redundant y reads)
__global__ __launch_bounds__(256)
void k_pre(const unsigned short* __restrict__ yb,
           const unsigned short* __restrict__ wihb,
           const float* __restrict__ xbih, const float* __restrict__ xbhh,
           const float* __restrict__ zbih, const float* __restrict__ zbhh,
           float* __restrict__ pre) {
  int combo = blockIdx.y;                  // e*2 + d
  int lane = threadIdx.x & 31, wave = threadIdx.x >> 5;
  int pairId = blockIdx.x * 8 + wave;      // 0..4095
  int mt = pairId >> 3;                    // 0..511
  int nt0 = (pairId & 7) * 2;              // 0,2,...,14
  const unsigned short* W = wihb + (size_t)combo * H_ * IN_;
  ABFrag a, b0, b1; v8f acc0 = {}, acc1 = {};
#pragma unroll
  for (int kk = 0; kk < IN_; kk += 32) {
    load_frag(a,  yb, mt * 16,        IN_, kk, lane);
    load_frag(b0, W,  nt0 * 16,       IN_, kk, lane);
    load_frag(b1, W,  nt0 * 16 + 16,  IN_, kk, lane);
    acc0 = wmma_bf16(a, b0, acc0);
    acc1 = wmma_bf16(a, b1, acc1);
  }
  int e = combo >> 1, d = combo & 1;
  int grp = lane >> 4;
  const float* bih = (e ? zbih : xbih) + d * H_;
  const float* bhh = (e ? zbhh : xbhh) + d * H_;
  float* outp = pre + (size_t)combo * ROWS_ * H_;
#pragma unroll
  for (int j = 0; j < 2; ++j) {
    int col = (nt0 + j) * 16 + (lane & 15);
    float bias = bih[col] + bhh[col];
#pragma unroll
    for (int r = 0; r < 8; ++r) {
      int row = mt * 16 + r + 8 * grp;
      float v = (j ? acc1[r] : acc0[r]) + bias;
      outp[(size_t)row * H_ + col] = v;
    }
  }
}

// --------- recurrent scan: one block per (encoder,direction) chain ---------
// Whh pinned in LDS (bf16, ld=264: b128 row-lane reads hit disjoint bank
// quads), h double-buffered in LDS. Each wave owns a 16x32 strip (2 N-tiles
// sharing A) -> 768KB LDS read/step instead of 1MB.
__device__ __forceinline__ void rnn_epilogue(v8f acc, int mt, int nt, int lane,
                                             int s, const float* __restrict__ prep,
                                             unsigned short* nxt,
                                             unsigned short* __restrict__ encp) {
  int col = nt * 16 + (lane & 15), grp = lane >> 4;
#pragma unroll
  for (int r = 0; r < 8; ++r) {
    int bb = mt * 16 + r + 8 * grp;
    float v = tanhf(acc[r] + prep[((size_t)bb * S_ + s) * H_ + col]);
    unsigned short bv = f2bf(v);
    nxt[bb * 264 + col] = bv;
    encp[((size_t)bb * S_ + s) * (2 * H_) + col] = bv;
  }
}

__global__ __launch_bounds__(1024)
void k_rnn(const unsigned short* __restrict__ whhb,
           const float* __restrict__ pre,
           unsigned short* __restrict__ encb) {
  extern __shared__ unsigned short smem[];
  unsigned short* sW = smem;                       // [256][264]
  unsigned short* sH = smem + 256 * 264;           // [2][64][264]
  const int combo = blockIdx.x, e = combo >> 1, d = combo & 1;
  const int tid = threadIdx.x, lane = tid & 31, wave = tid >> 5;

  const unsigned short* Wg = whhb + (size_t)combo * H_ * H_;
  // Stage Whh 256x256 bf16 -> LDS [256][264] in 16B chunks.
#if HAVE_ASYNC_LDS
  for (int i = tid; i < 8192; i += 1024) {
    int r = i >> 5, c = (i & 31) * 8;
    __builtin_amdgcn_global_load_async_to_lds_b128(
        (as1_v4i_p)(Wg + (size_t)r * 256 + c),
        (as3_v4i_p)(sW + r * 264 + c),
        0, 0);
  }
  __builtin_amdgcn_s_wait_asynccnt(0);
#else
  for (int i = tid; i < 8192; i += 1024) {
    int r = i >> 5, c = (i & 31) * 8;
    *(uint4*)(sW + r * 264 + c) = *(const uint4*)(Wg + (size_t)r * 256 + c);
  }
#endif
  for (int i = tid; i < 64 * 264; i += 1024) sH[i] = 0;   // h0 = 0
  __syncthreads();

  const float* prep = pre + (size_t)combo * ROWS_ * H_;
  unsigned short* encp = encb + (size_t)e * ROWS_ * (2 * H_) + d * H_;

  const int mt  = wave >> 3;          // 0..3   (batch tile)
  const int nt0 = (wave & 7) * 2;     // 0..14  (pair of col tiles)

  for (int t = 0; t < S_; ++t) {
    int s = d ? (S_ - 1 - t) : t;
    const unsigned short* cur = sH + (size_t)(t & 1) * 64 * 264;
    unsigned short* nxt = sH + (size_t)((t + 1) & 1) * 64 * 264;

    // prefetch next step's pre-activations (global_prefetch_b8)
    if (t + 1 < S_) {
      int sn = d ? (S_ - 2 - t) : (t + 1);
      __builtin_prefetch(prep + ((size_t)(tid >> 4) * S_ + sn) * H_ + (tid & 15) * 16, 0, 1);
    }

    ABFrag a, b0, b1; v8f acc0 = {}, acc1 = {};
#pragma unroll
    for (int kk = 0; kk < H_; kk += 32) {
      load_frag(a,  cur, mt * 16,       264, kk, lane);
      load_frag(b0, sW,  nt0 * 16,      264, kk, lane);
      load_frag(b1, sW,  nt0 * 16 + 16, 264, kk, lane);
      acc0 = wmma_bf16(a, b0, acc0);
      acc1 = wmma_bf16(a, b1, acc1);
    }
    rnn_epilogue(acc0, mt, nt0,     lane, s, prep, nxt, encp);
    rnn_epilogue(acc1, mt, nt0 + 1, lane, s, prep, nxt, encp);
    __syncthreads();
  }
}

// --------- output projection: mbd = enc * [Wm;Wbd]^T + [bm;bbd] ------------
__global__ __launch_bounds__(256)
void k_post(const unsigned short* __restrict__ encb,
            const unsigned short* __restrict__ wpost,
            const float* __restrict__ bm, const float* __restrict__ bbd,
            int n1, int n2, int Ntiles, int Ld, float* __restrict__ mbd) {
  int lane = threadIdx.x & 31, wave = threadIdx.x >> 5;
  int tile = blockIdx.x * 8 + wave;
  int mt = tile / Ntiles, nt = tile % Ntiles;
  ABFrag a, b; v8f acc = {};
#pragma unroll
  for (int kk = 0; kk < 2 * H_; kk += 32) {
    load_frag(a, encb,  mt * 16, 2 * H_, kk, lane);
    load_frag(b, wpost, nt * 16, 2 * H_, kk, lane);
    acc = wmma_bf16(a, b, acc);
  }
  int col = nt * 16 + (lane & 15), grp = lane >> 4;
  float bias = (col < n1) ? bm[col] : ((col < n1 + n2) ? bbd[col - n1] : 0.f);
#pragma unroll
  for (int r = 0; r < 8; ++r) {
    int row = mt * 16 + r + 8 * grp;
    mbd[(size_t)row * Ld + col] = acc[r] + bias;
  }
}

// --- tridiagonal Cholesky + backward bidiagonal solve, 1 thread per chain ---
// mbd row (b*S+s): [0..D) mean, [D..2D) a, [2D..3D) boff
__global__ void k_chol(const float* __restrict__ mbd, int Ld, int D, int NCH,
                       const float* __restrict__ eps, float* __restrict__ out,
                       float* __restrict__ Lb, float* __restrict__ Mb) {
  int chain = blockIdx.x * blockDim.x + threadIdx.x;
  if (chain >= NCH) return;
  int b = chain / D, d = chain % D;

  float a_prev = mbd[((size_t)b * S_ + 0) * Ld + D + d];
  float l = sqrtf(a_prev * a_prev + EPSV);
  Lb[chain] = l;
  for (int s = 0; s < S_ - 1; ++s) {
    float bo = mbd[((size_t)b * S_ + s) * Ld + 2 * D + d];
    float an = mbd[((size_t)b * S_ + s + 1) * Ld + D + d];
    float m  = (a_prev * bo) / l;
    float ln = sqrtf(an * an + bo * bo + EPSV - m * m);
    Mb[(size_t)s * NCH + chain] = m;
    Lb[(size_t)(s + 1) * NCH + chain] = ln;
    a_prev = an; l = ln;
  }
  for (int n = 0; n < NS_; ++n) {
    const float* ep = eps + ((size_t)(n * B_ + b) * S_) * D + d;
    float* op       = out + ((size_t)(n * B_ + b) * S_) * D + d;
    float x = ep[(size_t)(S_ - 1) * D] / Lb[(size_t)(S_ - 1) * NCH + chain];
    op[(size_t)(S_ - 1) * D] = mbd[((size_t)b * S_ + S_ - 1) * Ld + d] + x;
    for (int s = S_ - 2; s >= 0; --s) {
      x = (ep[(size_t)s * D] - Mb[(size_t)s * NCH + chain] * x)
          / Lb[(size_t)s * NCH + chain];
      op[(size_t)s * D] = mbd[((size_t)b * S_ + s) * Ld + d] + x;
    }
  }
}

// ------------------------------ softmax over Dz=3 ---------------------------
__global__ void k_softmax3(const float* __restrict__ zraw,
                           float* __restrict__ zout, int n) {
  int i = blockIdx.x * blockDim.x + threadIdx.x;
  if (i >= n) return;
  float a = zraw[i * 3], b = zraw[i * 3 + 1], c = zraw[i * 3 + 2];
  float mx = fmaxf(a, fmaxf(b, c));
  float ea = __expf(a - mx), eb = __expf(b - mx), ec = __expf(c - mx);
  float inv = 1.f / (ea + eb + ec);
  zout[i * 3] = ea * inv; zout[i * 3 + 1] = eb * inv; zout[i * 3 + 2] = ec * inv;
}

// --------- reconstruction: y = softplus(sum_k z_k (xg_k . Wl_k,i + bl)) -----
__global__ __launch_bounds__(256)
void k_recon(const float* __restrict__ Wl, const float* __restrict__ bl,
             const float* __restrict__ xs, const float* __restrict__ zs,
             float* __restrict__ yout) {
  __shared__ float sWl[DZ_ * IN_ * G_];   // 48 KB
  __shared__ float sbl[DZ_ * IN_];
  __shared__ float sx[16 * DX_];
  __shared__ float sz[16 * DZ_];
  int tid = threadIdx.x;
  for (int i = tid; i < DZ_ * IN_ * G_; i += 256) sWl[i] = Wl[i];
  for (int i = tid; i < DZ_ * IN_;      i += 256) sbl[i] = bl[i];
  size_t pos0 = (size_t)blockIdx.x * 16;
  for (int i = tid; i < 16 * DX_; i += 256) sx[i] = xs[pos0 * DX_ + i];
  for (int i = tid; i < 16 * DZ_; i += 256) sz[i] = zs[pos0 * DZ_ + i];
  __syncthreads();
  int ch = tid;                            // output channel 0..255
  for (int t = 0; t < 16; ++t) {
    float acc = 0.f;
#pragma unroll
    for (int k = 0; k < DZ_; ++k) {
      float dot = sbl[k * IN_ + ch];
      const float* w = &sWl[(k * IN_ + ch) * G_];
      const float* x = &sx[t * DX_ + k * G_];
#pragma unroll
      for (int g = 0; g < G_; ++g) dot += w[g] * x[g];
      acc += sz[t * DZ_ + k] * dot;
    }
    float sp = acc > 20.f ? acc : log1pf(__expf(acc));
    yout[(pos0 + t) * IN_ + ch] = fmaxf(sp, 1e-10f);
  }
}

// ---------------------------------------------------------------------------
extern "C" void kernel_launch(void* const* d_in, const int* in_sizes, int n_in,
                              void* d_out, int out_size, void* d_ws, size_t ws_size,
                              hipStream_t stream) {
  (void)in_sizes; (void)n_in; (void)out_size; (void)ws_size;
  const float* y     = (const float*)d_in[0];
  const float* eps_x = (const float*)d_in[1];
  const float* eps_z = (const float*)d_in[2];
  const float* xWih  = (const float*)d_in[3];
  const float* xWhh  = (const float*)d_in[4];
  const float* xbih  = (const float*)d_in[5];
  const float* xbhh  = (const float*)d_in[6];
  const float* xWm   = (const float*)d_in[7];
  const float* xbm   = (const float*)d_in[8];
  const float* xWbd  = (const float*)d_in[9];
  const float* xbbd  = (const float*)d_in[10];
  const float* zWih  = (const float*)d_in[11];
  const float* zWhh  = (const float*)d_in[12];
  const float* zbih  = (const float*)d_in[13];
  const float* zbhh  = (const float*)d_in[14];
  const float* zWm   = (const float*)d_in[15];
  const float* zbm   = (const float*)d_in[16];
  const float* zWbd  = (const float*)d_in[17];
  const float* zbbd  = (const float*)d_in[18];
  const float* Wl    = (const float*)d_in[19];
  const float* bl    = (const float*)d_in[20];

  // ---- workspace layout (deterministic; ~65 MB total) ----
  char* w = (char*)d_ws; size_t o = 0;
  auto take = [&](size_t bytes) -> char* {
    char* p = w + o; o = (o + bytes + 255) & ~(size_t)255; return p;
  };
  unsigned short* yb   = (unsigned short*)take((size_t)ROWS_ * IN_ * 2);
  unsigned short* wihb = (unsigned short*)take((size_t)4 * H_ * IN_ * 2);
  unsigned short* whhb = (unsigned short*)take((size_t)4 * H_ * H_ * 2);
  unsigned short* wpx  = (unsigned short*)take((size_t)144 * 512 * 2);
  unsigned short* wpz  = (unsigned short*)take((size_t)16 * 512 * 2);
  float* pre  = (float*)take((size_t)4 * ROWS_ * H_ * 4);
  unsigned short* encb = (unsigned short*)take((size_t)2 * ROWS_ * 512 * 2);
  float* mbdx = (float*)take((size_t)ROWS_ * 144 * 4);
  float* mbdz = (float*)take((size_t)ROWS_ * 16 * 4);
  float* Lx   = (float*)take((size_t)S_ * B_ * DX_ * 4);
  float* Mx   = (float*)take((size_t)S_ * B_ * DX_ * 4);
  float* Lz   = (float*)take((size_t)S_ * B_ * DZ_ * 4);
  float* Mz   = (float*)take((size_t)S_ * B_ * DZ_ * 4);
  float* zraw = (float*)take((size_t)NS_ * B_ * S_ * DZ_ * 4);

  float* yrec = (float*)d_out;                                   // 8,388,608
  float* xout = (float*)d_out + (size_t)NS_ * B_ * S_ * IN_;     // 1,572,864
  float* zout = xout + (size_t)NS_ * B_ * S_ * DX_;              //    98,304

  auto cvt = [&](const float* src, unsigned short* dst, int n) {
    k_cvt_bf16<<<(n + 255) / 256, 256, 0, stream>>>(src, dst, n);
  };
  // bf16 conversions (weights are native [N][K] for the X*W^T GEMMs)
  cvt(y,    yb,   ROWS_ * IN_);
  cvt(xWih, wihb,                2 * H_ * IN_);
  cvt(zWih, wihb + 2 * H_ * IN_, 2 * H_ * IN_);
  cvt(xWhh, whhb,                2 * H_ * H_);
  cvt(zWhh, whhb + 2 * H_ * H_,  2 * H_ * H_);
  cvt(xWm,  wpx,            DX_ * 512);
  cvt(xWbd, wpx + DX_ * 512, 2 * DX_ * 512);
  cvt(zWm,  wpz,            DZ_ * 512);
  cvt(zWbd, wpz + DZ_ * 512, 2 * DZ_ * 512);

  // input projections for all 4 (encoder,direction) combos: 8192x256x256 each
  k_pre<<<dim3(512, 4), 256, 0, stream>>>(yb, wihb, xbih, xbhh, zbih, zbhh, pre);

  // recurrent scans: 4 blocks, Whh + double-buffered h resident in LDS
  size_t rnn_lds = (size_t)(256 * 264 + 2 * 64 * 264) * 2;   // 202,752 B
  k_rnn<<<4, 1024, rnn_lds, stream>>>(whhb, pre, encb);

  // output projections: x -> 8192x144 (9 N-tiles), z -> 8192x16 (1 N-tile)
  k_post<<<(512 * 9) / 8, 256, 0, stream>>>(encb, wpx, xbm, xbbd,
                                            DX_, 2 * DX_, 9, 144, mbdx);
  k_post<<<512 / 8, 256, 0, stream>>>(encb + (size_t)ROWS_ * 512, wpz, zbm, zbbd,
                                      DZ_, 2 * DZ_, 1, 16, mbdz);

  // Cholesky + solves + mean add
  k_chol<<<(B_ * DX_) / 256, 256, 0, stream>>>(mbdx, 144, DX_, B_ * DX_,
                                               eps_x, xout, Lx, Mx);
  k_chol<<<1, 256, 0, stream>>>(mbdz, 16, DZ_, B_ * DZ_, eps_z, zraw, Lz, Mz);

  // softmax over Dz, then mixture reconstruction + softplus
  k_softmax3<<<(NS_ * B_ * S_) / 256, 256, 0, stream>>>(zraw, zout, NS_ * B_ * S_);
  k_recon<<<(NS_ * B_ * S_) / 16, 256, 0, stream>>>(Wl, bl, xout, zout, yrec);
}
